// Tree3_41558103556310
// MI455X (gfx1250) — compile-verified, matrix operand order checked by reading, not run on hardware
//
#include <hip/hip_runtime.h>

typedef __attribute__((ext_vector_type(16))) _Float16 v16h;
typedef __attribute__((ext_vector_type(2)))  _Float16 h2;
typedef __attribute__((ext_vector_type(8)))  float    v8f;

__device__ __forceinline__ float sigmoidf_(float x) {
    // v_exp_f32 + v_rcp_f32 (hardware approx reciprocal) -- no IEEE div chain
    return __builtin_amdgcn_rcpf(1.0f + __expf(-x));
}

// Swap values between the two 16-lane halves of a wave32.
// v_permlanex16_b32 with identity selects = pure-VALU equivalent of shfl_xor(16).
__device__ __forceinline__ float xhalf_swap(float v) {
#if __has_builtin(__builtin_amdgcn_permlanex16)
    const int r = __builtin_amdgcn_permlanex16(
        __float_as_int(v), __float_as_int(v),
        0x76543210, (int)0xfedcba98u, false, false);
    return __int_as_float(r);
#else
    return __shfl_xor(v, 16, 32);
#endif
}

// Per-lane K striping of the 16-bit 16x32 A operand (ISA 7.12.2):
//   half h (lane < 16):  K = h   (h<8) | h+8  (h>=8)
//   half h (lane >= 16): K = h+8 (h<8) | h+16 (h>=8)
// im2col offset for patch element K (K = 5*r + c): off = r*32 + c ; K>=25 -> 0
// (K>=25 has a zero B column, so any finite A value is fine.)
__device__ __constant__ const int OFF_LO[16] =
    {0, 1, 2, 3, 4, 32, 33, 34, 97, 98, 99, 100, 128, 129, 130, 131};
__device__ __constant__ const int OFF_HI[16] =
    {35, 36, 64, 65, 66, 67, 68, 96, 132, 0, 0, 0, 0, 0, 0, 0};

// One workgroup (256 threads = 8 wave32) per image.
// LDS-resident dataflow: image(f16) -> WMMA conv -> sigmoid+pool(LDS) -> tree -> FC.
__global__ __launch_bounds__(256)
void tree3_fused_kernel(const float* __restrict__ x,       // (B,3,32,32)
                        const float* __restrict__ conv_w,  // (45,1,5,5)
                        const float* __restrict__ conv_b,  // (45,)
                        const float* __restrict__ tree_w,  // (16,15,3,7,7,2,2)
                        const float* __restrict__ tree_b,  // (16,3,7)
                        const float* __restrict__ fc_w,    // (10,336)
                        const float* __restrict__ fc_b,    // (10,)
                        float* __restrict__ out)           // (B,10)
{
    __shared__ _Float16 xs[3 * 32 * 32 + 16];            // image (f16) + overrun pad
    __shared__ _Float16 wks[45 * 25];                    // conv weights (f16)
    __shared__ float    cbs[48];                         // conv bias, zero-padded
    __shared__ __align__(16) float pooled[45 * 14 * 14]; // pooled activations
    __shared__ __align__(16) float tact[336];            // tree activations

    const int b    = blockIdx.x;
    const int tid  = threadIdx.x;
    const int lane = tid & 31;
    const int wave = tid >> 5;
    const bool hi  = (lane >= 16);
    const int N    = lane & 15;  // channel within group (B/D layout: lane owns column N)

    // ---------- Phase 0: stage image + conv weights into LDS (f32 -> f16) ----------
    const float* xi = x + (size_t)b * 3072;
    for (int i = tid * 4; i < 3072; i += 1024) {          // float4 loads, packed h2 stores
        const float4 v = *(const float4*)(xi + i);
        h2 p0; p0[0] = (_Float16)v.x; p0[1] = (_Float16)v.y;
        h2 p1; p1[0] = (_Float16)v.z; p1[1] = (_Float16)v.w;
        *(h2*)(xs + i)     = p0;
        *(h2*)(xs + i + 2) = p1;
    }
    if (tid < 16) xs[3072 + tid] = (_Float16)0.0f;        // zero the pad tail
    for (int i = tid; i < 45 * 25; i += 256) wks[i] = (_Float16)conv_w[i];
    {
        float v = 0.0f;
        if (tid < 45) v = conv_b[tid];
        if (tid < 48) cbs[tid] = v;                       // zero-padded -> no guard later
    }
    // Warm L2/L0 for phase-2 weights while phase 1 runs (global_prefetch_b8).
    for (int o = tid; o < 336; o += 256) {
        const int f = o / 21, r = o % 21;
        __builtin_prefetch(tree_w + f * 8820 + (r / 7) * 196 + (r % 7) * 28, 0, 1);
    }
    __syncthreads();

    // ---------- Build B operands (K=32 x N=16, zero padded) per group ----------
    v16h wB[3];
#pragma unroll
    for (int g = 0; g < 3; ++g) {
#pragma unroll
        for (int h = 0; h < 16; ++h) {
            const int K = (h < 8 ? h : h + 8) + (hi ? 8 : 0);
            _Float16 v = (_Float16)0.0f;
            if (K < 25 && N < 15) v = wks[(g * 15 + N) * 25 + K];
            wB[g][h] = v;
        }
    }

    // ---------- Phase 1: conv(5x5) via WMMA + bias + sigmoid + 2x2 maxpool ----------
    // M-tile = 2 rows x 8 cols of conv-output pixels (pool closes inside the tile).
    // 14 row-pairs x 4 col-octets = 56 tiles/group = 8 waves x 7 -> uniform EXEC.
    const bool storer = (!hi) && (N < 15);                // pooled-store lanes
#pragma unroll
    for (int g = 0; g < 3; ++g) {
        const float bias = cbs[g * 15 + N];               // unconditional, hoisted
        for (int t = wave; t < 56; t += 8) {
            const int yp = t >> 2;                        // pooled row 0..13  (wave-uniform)
            const int xt = t & 3;                         // col octet 0..3    (wave-uniform)
            const int M  = lane & 15;                     // A layout: lane owns row M
            const int oy = 2 * yp + (M >> 3);             // conv row (<= 27)
            const int ox = 8 * xt + (M & 7);              // conv col (overhang discarded)
            const int pb = g * 1024 + oy * 32 + ox;

            v16h a;                                       // 16 unconditional ds_load_u16
#pragma unroll
            for (int h = 0; h < 16; ++h) {
                const int off = hi ? OFF_HI[h] : OFF_LO[h];   // v_cndmask of constants
                a[h] = xs[pb + off];
            }

            v8f acc = {};
            acc = __builtin_amdgcn_wmma_f32_16x16x32_f16(
                false, a, false, wB[g], (short)0, acc, false, false);

            // D layout: lane = channel N; acc[v] = pixel (row = hi, col = v) of the tile.
            float p[4];
#pragma unroll
            for (int k = 0; k < 4; ++k) {
                const float s0 = sigmoidf_(acc[2 * k]     + bias);
                const float s1 = sigmoidf_(acc[2 * k + 1] + bias);
                const float m0 = fmaxf(s0, s1);           // pool across the column pair
                p[k] = fmaxf(m0, xhalf_swap(m0));         // pool across the row pair (VALU)
            }
            if (storer) {                                 // two 8B-aligned b64 stores
                float* row = pooled + (g * 15 + N) * 196 + yp * 14 + 4 * xt;
                *(float2*)row = make_float2(p[0], p[1]);  // cols 4xt,4xt+1 always valid
                if (xt < 3)                               // scalar branch (xt uniform)
                    *(float2*)(row + 2) = make_float2(p[2], p[3]);
            }
        }
    }
    __syncthreads();

    // ---------- Phase 2: tree layer: out[f,g,i] = sum_{m,j,ki,kj} pooled * tree_w ----------
    // tree_w strides (f,m,g,i,j,ki,kj) = (8820, 588, 196, 28, 4, 2, 1); (ki,kj) quad = float4
    for (int o = tid; o < 336; o += 256) {
        const int f = o / 21;
        const int r = o % 21;
        const int g = r / 7;
        const int i = r % 7;
        float acc = tree_b[o];                            // flat (f,g,i) == o
        const float* twf = tree_w + f * 8820 + g * 196 + i * 28;
        for (int m = 0; m < 15; ++m) {
            const float*  pch = pooled + (g * 15 + m) * 196 + (2 * i) * 14;
            const float4* tw4 = (const float4*)(twf + m * 588);
#pragma unroll
            for (int j = 0; j < 7; ++j) {
                const float4 w = tw4[j];                  // (ki,kj) quad, 16B aligned
                acc += pch[2 * j]          * w.x;
                acc += pch[2 * j + 1]      * w.y;
                acc += pch[14 + 2 * j]     * w.z;
                acc += pch[14 + 2 * j + 1] * w.w;
            }
        }
        tact[o] = sigmoidf_(acc);
    }
    __syncthreads();

    // ---------- Phase 3: FC (10 x 336) ----------
    if (tid < 10) {
        float acc = fc_b[tid];
        const float4* w4 = (const float4*)(fc_w + tid * 336);
        const float4* t4 = (const float4*)tact;
#pragma unroll 4
        for (int k = 0; k < 84; ++k) {
            const float4 w = w4[k], t = t4[k];
            acc += t.x * w.x + t.y * w.y + t.z * w.z + t.w * w.w;
        }
        out[(size_t)b * 10 + tid] = acc;
    }
}

extern "C" void kernel_launch(void* const* d_in, const int* in_sizes, int n_in,
                              void* d_out, int out_size, void* d_ws, size_t ws_size,
                              hipStream_t stream) {
    (void)n_in; (void)d_ws; (void)ws_size; (void)out_size;
    const float* x      = (const float*)d_in[0];
    const float* conv_w = (const float*)d_in[1];
    const float* conv_b = (const float*)d_in[2];
    const float* tree_w = (const float*)d_in[3];
    const float* tree_b = (const float*)d_in[4];
    const float* fc_w   = (const float*)d_in[5];
    const float* fc_b   = (const float*)d_in[6];
    float* out = (float*)d_out;

    const int B = in_sizes[0] / (3 * 32 * 32);   // 2048
    tree3_fused_kernel<<<B, 256, 0, stream>>>(x, conv_w, conv_b, tree_w, tree_b,
                                              fc_w, fc_b, out);
}
